// SelfAttention_39324720562589
// MI455X (gfx1250) — matrix operand order repaired and checked by reading, not compiled, not run on hardware
//
#include <hip/hip_runtime.h>

// ---------------------------------------------------------------------------
// Single-head causal self-attention for gfx1250 (MI455X), flash-attention
// style. All matmuls on v_wmma_f32_16x16x32_f16 (f32 accumulate); all bulk
// data movement via global_load_async_to_lds_b128 (ASYNCcnt) with double
// buffering; WMMA operands read from LDS as contiguous f16 pairs.
//   B=2, S=4096, E=1024, D=64
// Pass 0: W^T f16 prepass (3 x [64][1024], 384 KB).
// Pass 1: Q = xWq, K = xWk, V^T = (xWv)^T stored f16 in workspace.
// Pass 2: fused flash attention, 8 waves/block sharing pipelined K/V stages.
// ---------------------------------------------------------------------------

typedef __attribute__((ext_vector_type(16))) _Float16 v16h;
typedef __attribute__((ext_vector_type(8)))  float    v8f;

namespace {
constexpr int kB = 2, kS = 4096, kE = 1024, kD = 64;
}

// ISA 7.12.2 16-bit A-matrix 16x32: lane(0-15)=row M, half selects K sub-block.
__device__ __forceinline__ int a_k(int v, int half) {
  return (v < 4 ? 0 : 16) + half * 8 + (v & 3) * 2;
}
// 16-bit B-matrix 32x16: lane(0-15)=col N; same K order as A (half selects
// K sub-block); 2 consecutive K per VGPR.
__device__ __forceinline__ int b_k(int v, int half) {
  return (v < 4 ? 0 : 16) + half * 8 + (v & 3) * 2;
}

// Async DMA: 16 bytes global -> LDS, tracked by ASYNCcnt (ISA 15.18.3 op 98).
__device__ __forceinline__ void async_b128(const void* g, const void* l) {
  unsigned lds_off = (unsigned)(uintptr_t)l;   // low 32 bits = LDS offset
  asm volatile("global_load_async_to_lds_b128 %0, %1, off"
               :: "v"(lds_off), "v"(g) : "memory");
}
__device__ __forceinline__ void wait_async0() {
  asm volatile("s_wait_asynccnt 0" ::: "memory");
}
__device__ __forceinline__ void wait_ds0() {
  asm volatile("s_wait_dscnt 0" ::: "memory");
}

// ---------------------------------------------------------------------------
// Pass 0: wt[which][d][e] = (f16) W[which][e][d]   (tiny: 768 KB read)
// ---------------------------------------------------------------------------
__global__ __launch_bounds__(128)
void wt_prep_kernel(const float* __restrict__ Wq,
                    const float* __restrict__ Wk,
                    const float* __restrict__ Wv,
                    _Float16* __restrict__ wt) {
  const int d     = blockIdx.x;            // 0..63
  const int which = blockIdx.y;            // 0..2
  const float* __restrict__ W = (which == 0) ? Wq : (which == 1) ? Wk : Wv;
  _Float16* __restrict__ o = wt + ((size_t)which * kD + d) * kE;
  for (int e = threadIdx.x; e < kE; e += 128)
    o[e] = (_Float16)W[(size_t)e * kD + d];
}

// ---------------------------------------------------------------------------
// Pass 1: 16-row tile per block; 4 waves own the four 16-col N tiles.
// Async-staged x chunk (f32, converted once to f16 in LDS) and W^T chunk
// (f16), double buffered across the 32 K-chunks. Inner loop: pure f16 LDS
// reads + one WMMA, zero conversions.
// ---------------------------------------------------------------------------
__global__ __launch_bounds__(128)
void qkv_proj_kernel(const float* __restrict__ x,
                     const _Float16* __restrict__ wt,
                     _Float16* __restrict__ q,
                     _Float16* __restrict__ kk_,
                     _Float16* __restrict__ vt) {
  const int tile  = blockIdx.x;            // row tile over flattened [B*S]
  const int which = blockIdx.y;            // 0=Q 1=K 2=V
  const _Float16* __restrict__ Wtg = wt + (size_t)which * kD * kE;

  const int t    = threadIdx.x;
  const int wave = t >> 5;
  const int lane = t & 31;
  const int half = lane >> 4;
  const int l15  = lane & 15;
  const int row0 = tile * 16;

  __shared__ float    xs[2][16 * 32];      // staged x chunk (f32)
  __shared__ _Float16 xh[2][16 * 32];      // converted x chunk (f16)
  __shared__ _Float16 wts[2][64 * 32];     // staged W^T chunk [dim][k] (f16)

  auto stage = [&](int kc, int buf) {
    // x[row0..+15][kc..+31]: 512 f32 = 128 threads x 1 b128
    async_b128(&x[(size_t)(row0 + (t >> 3)) * kE + kc + (t & 7) * 4],
               &xs[buf][(t >> 3) * 32 + (t & 7) * 4]);
    // W^T[0..63][kc..+31]: 2048 f16 = 128 threads x 2 b128
#pragma unroll
    for (int i = 0; i < 2; ++i) {
      const int e = (t * 2 + i) * 8;       // element in 64x32 chunk
      async_b128(&Wtg[(size_t)(e >> 5) * kE + kc + (e & 31)], &wts[buf][e]);
    }
  };

  stage(0, 0);
  v8f acc = {};
  for (int c = 0; c < kE / 32; ++c) {
    wait_async0();
    __syncthreads();                       // buffer c&1 staged for all waves
    if (c + 1 < kE / 32) stage((c + 1) * 32, (c + 1) & 1);

    // one-shot cooperative f32 -> f16 conversion of the x chunk
    {
      const float* __restrict__ xf = xs[c & 1];
      _Float16* __restrict__ xo = xh[c & 1];
#pragma unroll
      for (int i = 0; i < 4; ++i) xo[t + i * 128] = (_Float16)xf[t + i * 128];
    }
    __syncthreads();                       // f16 chunk visible to all waves

    const _Float16* __restrict__ xb = xh[c & 1];
    const _Float16* __restrict__ wb = wts[c & 1];
    v16h a, b;
#pragma unroll
    for (int vv = 0; vv < 8; ++vv) {
      const int ka = a_k(vv, half);
      a[2 * vv]     = xb[l15 * 32 + ka];
      a[2 * vv + 1] = xb[l15 * 32 + ka + 1];
      const int kb = b_k(vv, half);
      b[2 * vv]     = wb[(wave * 16 + l15) * 32 + kb];
      b[2 * vv + 1] = wb[(wave * 16 + l15) * 32 + kb + 1];
    }
    acc = __builtin_amdgcn_wmma_f32_16x16x32_f16(false, a, false, b,
                                                 (short)0, acc, false, false);
  }

  // C layout: VGPR vv -> row vv + 8*half, col = l15.
  if (which == 2) {
    // store V transposed: vt[bat][dim][S]
    const int bat = row0 / kS;             // tiles never straddle batches
    const int sr0 = row0 - bat * kS;
#pragma unroll
    for (int vv = 0; vv < 8; ++vv)
      vt[((size_t)bat * kD + wave * 16 + l15) * kS + sr0 + vv + 8 * half] =
          (_Float16)acc[vv];
  } else {
    _Float16* __restrict__ out = (which == 0) ? q : kk_;
#pragma unroll
    for (int vv = 0; vv < 8; ++vv)
      out[(size_t)(row0 + vv + 8 * half) * kD + wave * 16 + l15] =
          (_Float16)acc[vv];
  }
}

// ---------------------------------------------------------------------------
// Pass 2: flash attention. 256 threads = 8 waves, one 16-query tile each;
// shared double-buffered async staging of 32-key K and V^T blocks.
// ---------------------------------------------------------------------------
__global__ __launch_bounds__(256)
void flash_attn_kernel(const _Float16* __restrict__ Q,
                       const _Float16* __restrict__ K,
                       const _Float16* __restrict__ Vt,
                       float* __restrict__ out) {
  const int bx   = blockIdx.x;             // 128-query slab
  const int bat  = blockIdx.y;
  const int t    = threadIdx.x;
  const int wave = t >> 5;
  const int lane = t & 31;
  const int half = lane >> 4;
  const int l15  = lane & 15;
  const int i0   = bx * 128 + wave * 16;   // this wave's query tile

  const _Float16* __restrict__ Qb = Q  + (size_t)bat * kS * kD;
  const _Float16* __restrict__ Kb = K  + (size_t)bat * kS * kD;
  const _Float16* __restrict__ Vb = Vt + (size_t)bat * kD * kS;

  __shared__ _Float16 Kl[2][32 * 64];      // key-major [key][dim]
  __shared__ _Float16 Vl[2][64 * 32];      // dim-major [dim][key]
  __shared__ _Float16 Pl[8][16 * 32];      // per-wave P transpose staging

  auto stage = [&](int j0, int buf) {
    // K block: 2048 f16 = 256 threads x 1 b128
    async_b128(&Kb[(size_t)(j0 + (t >> 3)) * kD + (t & 7) * 8],
               &Kl[buf][(t >> 3) * 64 + (t & 7) * 8]);
    // V^T block: 2048 f16 = 256 threads x 1 b128
    async_b128(&Vb[(size_t)(t >> 2) * kS + j0 + (t & 3) * 8],
               &Vl[buf][(t >> 2) * 32 + (t & 3) * 8]);
  };

  // Q tile, A layout, two K-depth chunks (contiguous pairs in memory).
  v16h qa[2];
#pragma unroll
  for (int c = 0; c < 2; ++c)
#pragma unroll
    for (int vv = 0; vv < 8; ++vv) {
      const int d = c * 32 + a_k(vv, half);
      qa[c][2 * vv]     = Qb[(size_t)(i0 + l15) * kD + d];
      qa[c][2 * vv + 1] = Qb[(size_t)(i0 + l15) * kD + d + 1];
    }

  v8f   o0 = {}, o1 = {}, o2 = {}, o3 = {};
  float m[8], ll[8];
#pragma unroll
  for (int vv = 0; vv < 8; ++vv) { m[vv] = -INFINITY; ll[vv] = 0.0f; }

  const float scale = 0.125f;              // 1/sqrt(64)
  const int nj = 4 * (bx + 1);             // 32-key blocks for the whole slab

  stage(0, 0);
  for (int jb = 0; jb < nj; ++jb) {
    const int j0 = jb * 32;
    wait_async0();
    __syncthreads();                       // buffer jb&1 ready for all waves
    if (jb + 1 < nj) stage((jb + 1) * 32, (jb + 1) & 1);

    if (j0 <= i0 + 15) {                   // causal frontier for this wave
      const _Float16* __restrict__ kb = Kl[jb & 1];
      const _Float16* __restrict__ vl = Vl[jb & 1];
      _Float16* __restrict__ pw = Pl[wave];

      // ---- scores: two 16x16 C tiles ----
      v8f s0 = {}, s1 = {};
#pragma unroll
      for (int c = 0; c < 2; ++c) {
        v16h k0, k1;
#pragma unroll
        for (int vv = 0; vv < 8; ++vv) {
          const int d = c * 32 + b_k(vv, half);    // B[d][t] = K[t][d]
          k0[2 * vv]     = kb[l15 * 64 + d];
          k0[2 * vv + 1] = kb[l15 * 64 + d + 1];
          k1[2 * vv]     = kb[(16 + l15) * 64 + d];
          k1[2 * vv + 1] = kb[(16 + l15) * 64 + d + 1];
        }
        s0 = __builtin_amdgcn_wmma_f32_16x16x32_f16(false, qa[c], false, k0,
                                                    (short)0, s0, false, false);
        s1 = __builtin_amdgcn_wmma_f32_16x16x32_f16(false, qa[c], false, k1,
                                                    (short)0, s1, false, false);
      }

      // ---- online softmax (C layout: row = vv + 8*half, col = l15) ----
#pragma unroll
      for (int vv = 0; vv < 8; ++vv) {
        const int r  = i0 + vv + 8 * half;
        const int t0 = j0 + l15;
        const int t1 = j0 + 16 + l15;
        float e0 = (t0 <= r) ? s0[vv] * scale : -INFINITY;
        float e1 = (t1 <= r) ? s1[vv] * scale : -INFINITY;

        float mx = fmaxf(e0, e1);
#pragma unroll
        for (int off = 1; off < 16; off <<= 1)
          mx = fmaxf(mx, __shfl_xor(mx, off, 32));

        const float mnew  = fmaxf(m[vv], mx);
        const float alpha = __expf(m[vv] - mnew);
        m[vv] = mnew;

        const float p0 = __expf(e0 - mnew);
        const float p1 = __expf(e1 - mnew);
        float rs = p0 + p1;
#pragma unroll
        for (int off = 1; off < 16; off <<= 1)
          rs += __shfl_xor(rs, off, 32);
        ll[vv] = ll[vv] * alpha + rs;

        o0[vv] *= alpha; o1[vv] *= alpha; o2[vv] *= alpha; o3[vv] *= alpha;

        const int rl = vv + 8 * half;
        pw[rl * 32 + l15]      = (_Float16)p0;
        pw[rl * 32 + 16 + l15] = (_Float16)p1;
      }

      wait_ds0();                          // LDS RAW within this wave

      // ---- P in A layout (16x32) ----
      v16h pa;
#pragma unroll
      for (int vv = 0; vv < 8; ++vv) {
        const int ka = a_k(vv, half);
        pa[2 * vv]     = pw[l15 * 32 + ka];
        pa[2 * vv + 1] = pw[l15 * 32 + ka + 1];
      }

      // ---- O += P(16x32) x V(32x64): V^T gives contiguous pairs ----
      v16h vbop;
#pragma unroll
      for (int vv = 0; vv < 8; ++vv) {
        const int kk = b_k(vv, half);
        vbop[2 * vv]     = vl[(0 * 16 + l15) * 32 + kk];
        vbop[2 * vv + 1] = vl[(0 * 16 + l15) * 32 + kk + 1];
      }
      o0 = __builtin_amdgcn_wmma_f32_16x16x32_f16(false, pa, false, vbop,
                                                  (short)0, o0, false, false);
#pragma unroll
      for (int vv = 0; vv < 8; ++vv) {
        const int kk = b_k(vv, half);
        vbop[2 * vv]     = vl[(1 * 16 + l15) * 32 + kk];
        vbop[2 * vv + 1] = vl[(1 * 16 + l15) * 32 + kk + 1];
      }
      o1 = __builtin_amdgcn_wmma_f32_16x16x32_f16(false, pa, false, vbop,
                                                  (short)0, o1, false, false);
#pragma unroll
      for (int vv = 0; vv < 8; ++vv) {
        const int kk = b_k(vv, half);
        vbop[2 * vv]     = vl[(2 * 16 + l15) * 32 + kk];
        vbop[2 * vv + 1] = vl[(2 * 16 + l15) * 32 + kk + 1];
      }
      o2 = __builtin_amdgcn_wmma_f32_16x16x32_f16(false, pa, false, vbop,
                                                  (short)0, o2, false, false);
#pragma unroll
      for (int vv = 0; vv < 8; ++vv) {
        const int kk = b_k(vv, half);
        vbop[2 * vv]     = vl[(3 * 16 + l15) * 32 + kk];
        vbop[2 * vv + 1] = vl[(3 * 16 + l15) * 32 + kk + 1];
      }
      o3 = __builtin_amdgcn_wmma_f32_16x16x32_f16(false, pa, false, vbop,
                                                  (short)0, o3, false, false);
    }
  }

  // ---- epilogue: out = O / l ----
  float* __restrict__ ob = out + (size_t)bat * kS * kD;
#pragma unroll
  for (int vv = 0; vv < 8; ++vv) {
    const size_t row = (size_t)(i0 + vv + 8 * half);
    const float inv = 1.0f / ll[vv];
    ob[row * kD + 0 * 16 + l15] = o0[vv] * inv;
    ob[row * kD + 1 * 16 + l15] = o1[vv] * inv;
    ob[row * kD + 2 * 16 + l15] = o2[vv] * inv;
    ob[row * kD + 3 * 16 + l15] = o3[vv] * inv;
  }
}

// ---------------------------------------------------------------------------
extern "C" void kernel_launch(void* const* d_in, const int* in_sizes, int n_in,
                              void* d_out, int out_size, void* d_ws,
                              size_t ws_size, hipStream_t stream) {
  (void)in_sizes; (void)n_in; (void)out_size; (void)ws_size;
  const float* x  = (const float*)d_in[0];
  const float* Wq = (const float*)d_in[1];
  const float* Wk = (const float*)d_in[2];
  const float* Wv = (const float*)d_in[3];

  // workspace (f16): Q, K row-major; V^T dim-major; W^T (3.4 MB total)
  _Float16* q  = (_Float16*)d_ws;
  _Float16* k  = q  + (size_t)kB * kS * kD;
  _Float16* vt = k  + (size_t)kB * kS * kD;
  _Float16* wt = vt + (size_t)kB * kS * kD;

  dim3 g0(kD, 3), b0(128);
  wt_prep_kernel<<<g0, b0, 0, stream>>>(Wq, Wk, Wv, wt);

  dim3 g1(kB * kS / 16, 3), b1(128);
  qkv_proj_kernel<<<g1, b1, 0, stream>>>(x, wt, q, k, vt);

  dim3 g2(kS / 128, kB), b2(256);
  flash_attn_kernel<<<g2, b2, 0, stream>>>(q, k, vt, (float*)d_out);
}
